// DSVerifier_87058987090548
// MI455X (gfx1250) — compile-verified, matrix operand order checked by reading, not run on hardware
//
#include <hip/hip_runtime.h>

typedef __attribute__((ext_vector_type(16))) _Float16 v16h;
typedef __attribute__((ext_vector_type(8)))  float    v8f;

#define B_    100
#define C2_   128
#define C3_   256
#define N1    (B_ * C2_)        // 12800 c2 elements -> threads [0, 800)
#define N2    (B_ * C3_)        // 25600 c3 elements -> threads [800, 2400)
#define PIX1  (7 * 64 + 7)      // [7,7] inside 64x64 plane
#define PIX2  (3 * 32 + 3)      // [3,3] inside 32x32 plane
#define EPT   16                // elements per thread (fills one f16 A operand)
#define T_C2  (N1 / EPT)        // 800  (multiple of 32 -> wave-uniform split)
#define T_ALL ((N1 + N2) / EPT) // 2400

__global__ void dsv_zero_out(float* __restrict__ out) {
    if (threadIdx.x == 0) out[0] = 0.0f;
}

__global__ __launch_bounds__(160)
void dsv_gather_wmma_reduce(const float* __restrict__ c2,
                            const float* __restrict__ c3,
                            const float* __restrict__ mask1,
                            const float* __restrict__ mask2,
                            const float* __restrict__ med1p,
                            const float* __restrict__ med2p,
                            float* __restrict__ out)
{
    const int t = blockIdx.x * blockDim.x + threadIdx.x;

    float pix[EPT];
    float msk[EPT];
    float med;

    // Wave-uniform branch: t<800 is a whole number of waves, so EXEC stays
    // all-1s in each arm and every arm issues its 16 scattered pixel loads
    // (one VGPR base + immediate offsets) plus 4x b128 mask loads as one
    // clause before any use -> full per-thread MLP.
    if (t < T_C2) {
        med = med1p[0];
        const long base = (long)t * EPT;                 // (b*C2 + c) index
        const float* p  = c2 + base * 4096L + PIX1;      // pixel stride 4096 f
#pragma unroll
        for (int k = 0; k < EPT; ++k) pix[k] = p[(long)k * 4096L];
        const float4* m4 = (const float4*)(mask1 + base); // 64B-aligned
#pragma unroll
        for (int k = 0; k < EPT / 4; ++k) {
            const float4 v = m4[k];
            msk[4 * k + 0] = v.x; msk[4 * k + 1] = v.y;
            msk[4 * k + 2] = v.z; msk[4 * k + 3] = v.w;
        }
    } else {
        med = med2p[0];
        const long base = (long)(t - T_C2) * EPT;        // (b*C3 + c) index
        const float* p  = c3 + base * 1024L + PIX2;      // pixel stride 1024 f
#pragma unroll
        for (int k = 0; k < EPT; ++k) pix[k] = p[(long)k * 1024L];
        const float4* m4 = (const float4*)(mask2 + base);
#pragma unroll
        for (int k = 0; k < EPT / 4; ++k) {
            const float4 v = m4[k];
            msk[4 * k + 0] = v.x; msk[4 * k + 1] = v.y;
            msk[4 * k + 2] = v.z; msk[4 * k + 3] = v.w;
        }
    }

    // diff^2 is exactly {0,1}: exact in f16, exact in f32 accumulation.
    v16h a;
#pragma unroll
    for (int k = 0; k < EPT; ++k) {
        const float tm = (pix[k] > med) ? 1.0f : 0.0f;
        const float d  = tm - msk[k];
        a[k] = (_Float16)(d * d);
    }

    // B = ones => every column of D = vector of row-sums of A; one column
    // (lane L plus lane L+16) sums all 512 values held by the wave.
    v16h ones;
#pragma unroll
    for (int k = 0; k < 16; ++k) ones[k] = (_Float16)1.0f;

    v8f acc = {};
    acc = __builtin_amdgcn_wmma_f32_16x16x32_f16(
        /*neg_a=*/false, a, /*neg_b=*/false, ones,
        /*c_mod=*/(short)0, acc, /*reuse_a=*/false, /*reuse_b=*/false);

    const float s = acc[0] + acc[1] + acc[2] + acc[3] +
                    acc[4] + acc[5] + acc[6] + acc[7];

    const int lane = threadIdx.x & 31;
    if (lane == 0 || lane == 16) {
        atomicAdd(out, s);   // integer-valued f32: order-independent, exact
    }
}

extern "C" void kernel_launch(void* const* d_in, const int* in_sizes, int n_in,
                              void* d_out, int out_size, void* d_ws, size_t ws_size,
                              hipStream_t stream)
{
    (void)in_sizes; (void)n_in; (void)out_size; (void)d_ws; (void)ws_size;

    const float* c2    = (const float*)d_in[0];
    const float* c3    = (const float*)d_in[1];
    const float* mask1 = (const float*)d_in[2];
    const float* mask2 = (const float*)d_in[3];
    const float* med1  = (const float*)d_in[4];
    const float* med2  = (const float*)d_in[5];
    float*       out   = (float*)d_out;

    // Deterministic across graph replays: re-zero the accumulator each launch.
    dsv_zero_out<<<1, 32, 0, stream>>>(out);

    // 15 blocks x 160 threads (5 waves) = exactly 2400 threads, no tail.
    dsv_gather_wmma_reduce<<<15, 160, 0, stream>>>(
        c2, c3, mask1, mask2, med1, med2, out);
}
